// SelfAttentionV1_77197742178886
// MI455X (gfx1250) — matrix operand order repaired and checked by reading, not compiled, not run on hardware
//
#include <hip/hip_runtime.h>

// ---------------------------------------------------------------------------
// SelfAttention (linear attention + 1x1/3x3 conv branches) for MI455X gfx1250.
// All GEMM-shaped work runs on v_wmma_f32_16x16x32_bf16 with fp32 accumulate.
// The dominant 3x3 conv stages its B-operand (3 image rows x 256ch, 96KB) in
// LDS via the Tensor Data Mover (tensor_load_to_lds + s_wait_tensorcnt), so
// all 8 waves of a workgroup share one 96KB DMA instead of ~2.3MB of per-wave
// L2 reads. Wave tile for big GEMMs: 32(M) x 64(N), 8 WMMAs per 32-K step.
// ---------------------------------------------------------------------------

typedef __bf16 bf16_t;
typedef __attribute__((ext_vector_type(8)))  bf16_t bf16x8;
typedef __attribute__((ext_vector_type(16))) bf16_t v16bf;
typedef __attribute__((ext_vector_type(8)))  float  v8f;
typedef __attribute__((ext_vector_type(8)))  unsigned short ushort8;
typedef __attribute__((ext_vector_type(4)))  unsigned int u32x4;
typedef __attribute__((ext_vector_type(8)))  int i32x8;
typedef __attribute__((ext_vector_type(4)))  int i32x4;

#define CAT16(lo, hi) __builtin_shufflevector((lo), (hi), \
    0, 1, 2, 3, 4, 5, 6, 7, 8, 9, 10, 11, 12, 13, 14, 15)

static constexpr int Bn = 8;      // batch
static constexpr int Cc = 256;    // channels
static constexpr int Nn = 4096;   // H*W
static constexpr int NT = Bn * Nn;
static constexpr int CQ = 32;     // query channels

__device__ __forceinline__ unsigned short f32_to_bf16(float f) {
  unsigned u = __float_as_uint(f);
  u += 0x7FFFu + ((u >> 16) & 1u);   // round-to-nearest-even
  return (unsigned short)(u >> 16);
}
__device__ __forceinline__ float bf16_to_f32(unsigned short h) {
  return __uint_as_float(((unsigned)h) << 16);
}

// A-operand: two separated 8-element chunks (each 16B aligned).
__device__ __forceinline__ v16bf load_a16(const unsigned short* p0,
                                          const unsigned short* p1) {
  return CAT16(*(const bf16x8*)p0, *(const bf16x8*)p1);
}
// B-operand: 16 contiguous bf16 (32B).
__device__ __forceinline__ v16bf load_b16(const unsigned short* p) {
  const bf16x8* q = (const bf16x8*)p;
  return CAT16(q[0], q[1]);
}
__device__ __forceinline__ v8f wmma_bf16(v16bf a, v16bf b, v8f c) {
  return __builtin_amdgcn_wmma_f32_16x16x32_bf16(
      false, a, false, b, (short)0, c, false, false);
}

// Raw LDS byte offset of a shared-memory pointer (generic -> AS3 -> int).
__device__ __forceinline__ unsigned lds_offset_of(const void* p) {
  return (unsigned)(unsigned long long)
      (__attribute__((address_space(3))) const void*)p;
}

// Issue one TDM load: `rows` lines of 256 bf16 (contiguous, stride 256) from
// global byte address ga into LDS byte offset laddr.  D# per ISA 8.3/8.4.
// This toolchain exposes the 6-arg builtin: (g0, g1, g2, g3, g4, cpol).
__device__ __forceinline__ void tdm_load_rows(unsigned long long ga,
                                              unsigned laddr, int rows) {
  u32x4 g0;
  g0[0] = 1u;                                   // count=1, user descriptor
  g0[1] = laddr;                                // LDS byte address
  g0[2] = (unsigned)ga;                         // global addr [31:0]
  g0[3] = ((unsigned)(ga >> 32) & 0x01FFFFFFu)  // global addr [56:32]
          | 0x80000000u;                        // type = 2 ("image")
  i32x8 g1;
  g1[0] = 0x00010000;        // workgroup_mask=0, data_size=1 (2 bytes)
  g1[1] = 0x01000000;        // tensor_dim0 = 256 (low 16 of bits 79:48)
  g1[2] = (int)0x80000000u;  // tensor_dim1 = 32768 (low 16 of bits 111:80)
  g1[3] = 0x01000000;        // tile_dim0 = 256 (bits 127:112)
  g1[4] = rows;              // tile_dim1 = rows, tile_dim2 = 0
  g1[5] = 256;               // tensor_dim0_stride = 256 (data_size units)
  g1[6] = 0;
  g1[7] = 0;
  i32x4 gz4 = {0, 0, 0, 0};                     // groups 2/3 unused (2D)
  i32x8 gz8 = {0, 0, 0, 0, 0, 0, 0, 0};
  __builtin_amdgcn_tensor_load_to_lds(g0, g1, gz4, gz4, gz8, 0);
}

// --------------------------- prep kernels ----------------------------------

// (B, R, 4096) f32  ->  (B, 4096, R) bf16, LDS-tiled 32x32 transpose.
__global__ void __launch_bounds__(256)
transpose_to_bf16_kernel(const float* __restrict__ in,
                         unsigned short* __restrict__ out, int R) {
  __shared__ float tile[32][33];
  int b  = blockIdx.z;
  int c0 = blockIdx.y * 32;
  int n0 = blockIdx.x * 32;
  int tx = threadIdx.x;   // 0..31
  int ty = threadIdx.y;   // 0..7
  const float* src = in + (size_t)b * R * Nn;
#pragma unroll
  for (int j = 0; j < 4; ++j)
    tile[ty + 8 * j][tx] = src[(size_t)(c0 + ty + 8 * j) * Nn + n0 + tx];
  __syncthreads();
  unsigned short* dst = out + (size_t)b * Nn * R;
#pragma unroll
  for (int j = 0; j < 4; ++j) {
    int n = n0 + ty + 8 * j;
    int c = c0 + tx;
    dst[(size_t)n * R + c] = f32_to_bf16(tile[tx][ty + 8 * j]);
  }
}

__global__ void __launch_bounds__(256)
f32_to_bf16_kernel(const float* __restrict__ in,
                   unsigned short* __restrict__ out, int n) {
  int i = blockIdx.x * 256 + threadIdx.x;
  if (i < n) out[i] = f32_to_bf16(in[i]);
}

// w3 (O,C,3,3) f32 -> w3r (O, 9*256) bf16 with k = tap*256 + c.
__global__ void __launch_bounds__(256)
reorder_w3_kernel(const float* __restrict__ w3,
                  unsigned short* __restrict__ out) {
  int i = blockIdx.x * 256 + threadIdx.x;
  if (i < Cc * 2304) {
    int o = i / 2304, k = i % 2304;
    int tap = k >> 8, c = k & 255;
    out[i] = f32_to_bf16(w3[(size_t)o * 2304 + c * 9 + tap]);
  }
}

// --------------------------- 1x1 conv GEMM ---------------------------------
// Y(O, NT) = W(O,256) * Xt^T. Wave tile 32(O) x 64(pix), K loop 256/32.
__global__ void __launch_bounds__(256)
gemm1x1_wmma_kernel(const unsigned short* __restrict__ Wbf,
                    const float* __restrict__ bias,
                    const unsigned short* __restrict__ Xt,
                    float* __restrict__ outF,
                    unsigned short* __restrict__ outB, int O) {
  int wave = (blockIdx.x * 256 + threadIdx.x) >> 5;
  int lane = threadIdx.x & 31;
  int half = lane >> 4, rc = lane & 15;
  int nChunks = NT / 64;                    // 512
  int pixChunk = wave % nChunks;
  int oGroup = wave / nChunks;              // 32 output rows per wave
  int o0 = oGroup * 32, p0 = pixChunk * 64;

  v8f acc[2][4] = {};
  const unsigned short* arow0 = Wbf + (size_t)(o0 + rc) * Cc;
  const unsigned short* arow1 = Wbf + (size_t)(o0 + 16 + rc) * Cc;
#pragma unroll
  for (int k0 = 0; k0 < Cc; k0 += 32) {
    v16bf A0 = load_a16(arow0 + k0 + half * 16, arow0 + k0 + half * 16 + 8);
    v16bf A1 = load_a16(arow1 + k0 + half * 16, arow1 + k0 + half * 16 + 8);
#pragma unroll
    for (int t = 0; t < 4; ++t) {
      int p = p0 + t * 16 + rc;
      v16bf Bm = load_b16(Xt + (size_t)p * Cc + k0 + half * 16);
      acc[0][t] = wmma_bf16(A0, Bm, acc[0][t]);
      acc[1][t] = wmma_bf16(A1, Bm, acc[1][t]);
    }
  }
  int b = p0 >> 12;
#pragma unroll
  for (int g = 0; g < 2; ++g) {
#pragma unroll
    for (int t = 0; t < 4; ++t) {
      int n = (p0 + t * 16 + rc) & (Nn - 1);
#pragma unroll
      for (int r = 0; r < 8; ++r) {
        int o = o0 + g * 16 + r + 8 * half;
        float v = acc[g][t][r] + bias[o];
        size_t idx = ((size_t)b * O + o) * Nn + n;
        if (outF) outF[idx] = v;
        if (outB) outB[idx] = f32_to_bf16(v);
      }
    }
  }
}

// --------------------------- 3x3 conv (implicit GEMM, TDM-staged) ----------
// One workgroup owns one image row (64 px) and all 256 output channels.
// Rows h-1..h+1 x 256ch (<=96KB) are staged into LDS with a single
// tensor_load_to_lds DMA; out-of-image rows are zero-filled. Each of the 8
// waves computes a 32(O) x 64(pix) tile; w+-1 edges are handled by predicated
// ds_load_b128 reads.
__global__ void __launch_bounds__(256)
conv3x3_wmma_lds_kernel(const unsigned short* __restrict__ w3r,
                        const float* __restrict__ b3,
                        const unsigned short* __restrict__ Xt,
                        float* __restrict__ d3) {
  extern __shared__ unsigned short smem[];   // 3 * 64 * 256 bf16 = 96KB
  int tid = threadIdx.x;
  int lane = tid & 31;
  int waveId = tid >> 5;                     // 0..7 -> output channel group
  int half = lane >> 4, rc = lane & 15;

  int chunk = blockIdx.x;                    // 0..511 : one image row
  int b = chunk >> 6;
  int h = chunk & 63;

  // ---- stage rows [h-1, h+1] into LDS slots 0..2 -------------------------
  int start = (h > 0) ? h - 1 : 0;
  int end = (h < 63) ? h + 1 : 63;
  int cnt = end - start + 1;                 // 2 or 3
  int slot0 = start - (h - 1);               // 0 or 1

  if (h == 0 || h == 63) {                   // zero-fill missing halo slot
    int s = (h == 0) ? 0 : 2;
    ushort8 z = {};
    ushort8* dst = (ushort8*)(smem + s * 16384);
    for (int i = tid; i < 2048; i += 256) dst[i] = z;
  }
  if (tid < 32) {                            // wave 0 issues the DMA
    unsigned long long ga =
        (unsigned long long)(const unsigned short*)(Xt +
            (size_t)(b * Nn + start * 64) * Cc);
    unsigned laddr = lds_offset_of(smem) + (unsigned)(slot0 * 32768);
    tdm_load_rows(ga, laddr, 64 * cnt);
    __builtin_amdgcn_s_wait_tensorcnt(0);
  }
  __syncthreads();

  // ---- compute ------------------------------------------------------------
  int o0 = waveId * 32;
  const unsigned short* arow0 = w3r + (size_t)(o0 + rc) * 2304;
  const unsigned short* arow1 = w3r + (size_t)(o0 + 16 + rc) * 2304;
  v8f acc[2][4] = {};

  for (int tap = 0; tap < 9; ++tap) {
    int dh = tap / 3 - 1, dw = tap % 3 - 1;
    int srcOff[4];
    bool ok[4];
#pragma unroll
    for (int t = 0; t < 4; ++t) {
      int w2 = t * 16 + rc + dw;
      ok[t] = (unsigned)w2 < 64u;
      srcOff[t] = ((dh + 1) * 64 + w2) * Cc;
    }
    if (tap < 8) {                           // prefetch next tap's weights
      int k2 = (tap + 1) * 256;
      __builtin_prefetch(arow0 + k2, 0, 1);
      __builtin_prefetch(arow1 + k2, 0, 1);
    }
    for (int c0 = 0; c0 < Cc; c0 += 32) {
      int k = tap * 256 + c0;
      v16bf A0 = load_a16(arow0 + k + half * 16, arow0 + k + half * 16 + 8);
      v16bf A1 = load_a16(arow1 + k + half * 16, arow1 + k + half * 16 + 8);
#pragma unroll
      for (int t = 0; t < 4; ++t) {
        bf16x8 lo = {}, hi = {};
        if (ok[t]) {
          const bf16x8* bp = (const bf16x8*)(smem + srcOff[t] + c0 + half * 16);
          lo = bp[0];
          hi = bp[1];
        }
        v16bf Bm = CAT16(lo, hi);
        acc[0][t] = wmma_bf16(A0, Bm, acc[0][t]);
        acc[1][t] = wmma_bf16(A1, Bm, acc[1][t]);
      }
    }
  }
#pragma unroll
  for (int g = 0; g < 2; ++g) {
#pragma unroll
    for (int t = 0; t < 4; ++t) {
      int n = h * 64 + t * 16 + rc;
#pragma unroll
      for (int r = 0; r < 8; ++r) {
        int o = o0 + g * 16 + r + 8 * half;
        d3[((size_t)b * Cc + o) * Nn + n] = acc[g][t][r] + b3[o];
      }
    }
  }
}

// --------------------------- attention pieces ------------------------------

// ksum[b][m] = sum_n k[b][m][n]
__global__ void __launch_bounds__(256)
ksum_kernel(const unsigned short* __restrict__ kbf, float* __restrict__ ksum) {
  int bm = blockIdx.x;                     // 0..255 = b*32+m
  const unsigned short* row = kbf + (size_t)bm * Nn;
  float s = 0.f;
  for (int i = threadIdx.x; i < Nn; i += 256) s += bf16_to_f32(row[i]);
  __shared__ float red[256];
  red[threadIdx.x] = s;
  __syncthreads();
  for (int off = 128; off > 0; off >>= 1) {
    if (threadIdx.x < off) red[threadIdx.x] += red[threadIdx.x + off];
    __syncthreads();
  }
  if (threadIdx.x == 0) ksum[bm] = red[0];
}

// norm[b][n] = 1 / sum_m qt[b][n][m] * (ksum[b][m] + EPS)
__global__ void __launch_bounds__(256)
norm_kernel(const unsigned short* __restrict__ qt,
            const float* __restrict__ ksum, float* __restrict__ nrm) {
  int p = blockIdx.x * 256 + threadIdx.x;  // 0..NT-1
  int b = p >> 12;
  const unsigned short* qrow = qt + (size_t)p * CQ;
  const float* ks = ksum + b * CQ;
  float acc = 0.f;
#pragma unroll
  for (int m = 0; m < CQ; ++m)
    acc += bf16_to_f32(qrow[m]) * (ks[m] + 1e-10f);
  nrm[p] = 1.0f / acc;
}

// kvT[b][c][m] = sum_n k[b][m][n] * v[b][c][n]   (K = 4096)
__global__ void __launch_bounds__(256)
kv_wmma_kernel(const unsigned short* __restrict__ kbf,   // (B,32,4096)
               const unsigned short* __restrict__ vbf,   // (B,256,4096)
               unsigned short* __restrict__ kvT) {       // (B,256,32)
  int wave = (blockIdx.x * 256 + threadIdx.x) >> 5;      // 256 waves
  int lane = threadIdx.x & 31;
  int half = lane >> 4, rc = lane & 15;
  int cTile = wave & 15;
  int mTile = (wave >> 4) & 1;
  int b = wave >> 5;
  int m0 = mTile * 16, c0 = cTile * 16;

  v8f acc = {};
  const unsigned short* arow = kbf + ((size_t)b * CQ + m0 + rc) * Nn;
  const unsigned short* brow = vbf + ((size_t)b * Cc + c0 + rc) * Nn;
  for (int n0 = 0; n0 < Nn; n0 += 32) {
    v16bf A = load_a16(arow + n0 + half * 16, arow + n0 + half * 16 + 8);
    v16bf Bm = load_b16(brow + n0 + half * 16);
    acc = wmma_bf16(A, Bm, acc);
  }
  ushort8 ov;
#pragma unroll
  for (int r = 0; r < 8; ++r) ov[r] = f32_to_bf16(acc[r]);
  *(ushort8*)(kvT + ((size_t)b * Cc + c0 + rc) * CQ + m0 + 8 * half) = ov;
}

// out = x + gamma * (attn*norm + 2*d1 + 2*d3), attn(c,n) = kvT(c,:)·qt(n,:)
__global__ void __launch_bounds__(256)
attn_final_kernel(const unsigned short* __restrict__ kvT,  // (B,256,32)
                  const unsigned short* __restrict__ qt,   // (B,4096,32)
                  const float* __restrict__ nrm,           // (B*4096)
                  const float* __restrict__ x,
                  const float* __restrict__ d1,
                  const float* __restrict__ d3,
                  const float* __restrict__ gamma,
                  float* __restrict__ out) {
  int wave = (blockIdx.x * 256 + threadIdx.x) >> 5;        // 8192 waves
  int lane = threadIdx.x & 31;
  int half = lane >> 4, rc = lane & 15;
  int chunk = wave & 63;            // 64 pixel-chunks per batch
  int cTile = (wave >> 6) & 15;
  int b = wave >> 10;
  int c0 = cTile * 16, n0 = chunk * 64;

  const unsigned short* arow = kvT + ((size_t)b * Cc + c0 + rc) * CQ;
  v16bf A = load_a16(arow + half * 16, arow + half * 16 + 8);
  float g = gamma[0];

#pragma unroll
  for (int t = 0; t < 4; ++t) {
    int n = n0 + t * 16 + rc;
    v16bf Bm = load_b16(qt + ((size_t)b * Nn + n) * CQ + half * 16);
    v8f acc = {};
    acc = wmma_bf16(A, Bm, acc);
    float nv = nrm[b * Nn + n];
#pragma unroll
    for (int r = 0; r < 8; ++r) {
      int c = c0 + r + 8 * half;
      size_t idx = ((size_t)b * Cc + c) * Nn + n;
      float attn = acc[r] * nv;
      out[idx] = x[idx] + g * (attn + 2.0f * d1[idx] + 2.0f * d3[idx]);
    }
  }
}

// --------------------------- launcher --------------------------------------

extern "C" void kernel_launch(void* const* d_in, const int* in_sizes, int n_in,
                              void* d_out, int out_size, void* d_ws,
                              size_t ws_size, hipStream_t stream) {
  (void)in_sizes; (void)n_in; (void)out_size; (void)ws_size;
  const float* x  = (const float*)d_in[0];
  const float* wq = (const float*)d_in[1];
  const float* bq = (const float*)d_in[2];
  const float* wk = (const float*)d_in[3];
  const float* bk = (const float*)d_in[4];
  const float* wv = (const float*)d_in[5];
  const float* bv = (const float*)d_in[6];
  const float* w1 = (const float*)d_in[7];
  const float* b1 = (const float*)d_in[8];
  const float* w3 = (const float*)d_in[9];
  const float* b3 = (const float*)d_in[10];
  const float* gm = (const float*)d_in[11];
  float* out = (float*)d_out;

  // Workspace carve (~111 MB), 256B aligned.
  char* base = (char*)d_ws;
  size_t off = 0;
  auto carve = [&](size_t bytes) -> char* {
    char* p = base + off;
    off += (bytes + 255) & ~(size_t)255;
    return p;
  };
  unsigned short* Xt   = (unsigned short*)carve((size_t)NT * Cc * 2);      // 16.8MB
  unsigned short* wq_b = (unsigned short*)carve((size_t)CQ * Cc * 2);
  unsigned short* wk_b = (unsigned short*)carve((size_t)CQ * Cc * 2);
  unsigned short* wv_b = (unsigned short*)carve((size_t)Cc * Cc * 2);
  unsigned short* w1_b = (unsigned short*)carve((size_t)Cc * Cc * 2);
  unsigned short* w3_b = (unsigned short*)carve((size_t)Cc * 2304 * 2);    // 1.2MB
  float*          q_f  = (float*)carve((size_t)Bn * CQ * Nn * 4);          // 4.2MB
  unsigned short* qt_b = (unsigned short*)carve((size_t)Bn * Nn * CQ * 2); // 2.1MB
  unsigned short* k_b  = (unsigned short*)carve((size_t)Bn * CQ * Nn * 2); // 2.1MB
  unsigned short* v_b  = (unsigned short*)carve((size_t)Bn * Cc * Nn * 2); // 16.8MB
  float*          d1f  = (float*)carve((size_t)Bn * Cc * Nn * 4);          // 33.5MB
  float*          d3f  = (float*)carve((size_t)Bn * Cc * Nn * 4);          // 33.5MB
  unsigned short* kvT  = (unsigned short*)carve((size_t)Bn * Cc * CQ * 2);
  float*          ksum = (float*)carve((size_t)Bn * CQ * 4);
  float*          nrm  = (float*)carve((size_t)NT * 4);

  dim3 tb(32, 8);
  // Prep: pixel-major bf16 activations + bf16 weights.
  transpose_to_bf16_kernel<<<dim3(Nn / 32, Cc / 32, Bn), tb, 0, stream>>>(x, Xt, Cc);
  f32_to_bf16_kernel<<<(CQ * Cc + 255) / 256, 256, 0, stream>>>(wq, wq_b, CQ * Cc);
  f32_to_bf16_kernel<<<(CQ * Cc + 255) / 256, 256, 0, stream>>>(wk, wk_b, CQ * Cc);
  f32_to_bf16_kernel<<<(Cc * Cc + 255) / 256, 256, 0, stream>>>(wv, wv_b, Cc * Cc);
  f32_to_bf16_kernel<<<(Cc * Cc + 255) / 256, 256, 0, stream>>>(w1, w1_b, Cc * Cc);
  reorder_w3_kernel<<<(Cc * 2304 + 255) / 256, 256, 0, stream>>>(w3, w3_b);

  // 1x1 conv GEMMs (WMMA). 32x64 wave tiles: waves = (O/32)*(NT/64).
  gemm1x1_wmma_kernel<<<64, 256, 0, stream>>>(wq_b, bq, Xt, q_f, nullptr, CQ);
  gemm1x1_wmma_kernel<<<64, 256, 0, stream>>>(wk_b, bk, Xt, nullptr, k_b, CQ);
  gemm1x1_wmma_kernel<<<512, 256, 0, stream>>>(wv_b, bv, Xt, nullptr, v_b, Cc);
  gemm1x1_wmma_kernel<<<512, 256, 0, stream>>>(w1_b, b1, Xt, d1f, nullptr, Cc);

  // Attention smalls.
  transpose_to_bf16_kernel<<<dim3(Nn / 32, 1, Bn), tb, 0, stream>>>(q_f, qt_b, CQ);
  ksum_kernel<<<Bn * CQ, 256, 0, stream>>>(k_b, ksum);
  norm_kernel<<<NT / 256, 256, 0, stream>>>(qt_b, ksum, nrm);
  kv_wmma_kernel<<<32, 256, 0, stream>>>(k_b, v_b, kvT);

  // 3x3 conv as implicit GEMM: one workgroup per image row, B staged in LDS
  // by the Tensor Data Mover (96KB dynamic LDS).
  conv3x3_wmma_lds_kernel<<<512, 256, 3 * 64 * Cc * 2, stream>>>(w3_b, b3, Xt, d3f);

  // Attention apply (WMMA, K=32) fused with final combine.
  attn_final_kernel<<<1024, 256, 0, stream>>>(kvT, qt_b, nrm, x, d1f, d3f, gm, out);
}
